// TopKRouter_18184891532040
// MI455X (gfx1250) — compile-verified, matrix operand order checked by reading, not run on hardware
//
#include <hip/hip_runtime.h>

typedef __attribute__((ext_vector_type(16))) __bf16 v16bf;
typedef __attribute__((ext_vector_type(8)))  __bf16 v8bf;
typedef __attribute__((ext_vector_type(4)))  __bf16 v4bf;
typedef __attribute__((ext_vector_type(8)))  float  v8f;
typedef __attribute__((ext_vector_type(4)))  float  v4f;

#define DMODEL 2048
#define NEXP   64
#define NTOK   16384
#define WAVES_PER_BLOCK 4
#define BLOCK_THREADS   128
#define ROWS_PER_WAVE   16
#define LDS_PITCH       65              // logits pitch: 64 + 1 pad float
#define KCHUNK          512             // K elements staged per chunk
#define BPITCH          (KCHUNK + 8)    // 520 bf16 = 1040B rows -> conflict-free ds_load_b128
#define GRANULES        (NEXP * (KCHUNK / 8))   // 16B async granules per chunk = 4096

// pack 16 fp32 (two groups of 8 contiguous K) into bf16 A-operand layout
__device__ __forceinline__ v16bf cvt16(v4f a0, v4f a1, v4f a2, v4f a3) {
  v16bf r;
#pragma unroll
  for (int i = 0; i < 4; ++i) {
    r[i]      = (__bf16)a0[i];   // K = k+0..3
    r[i + 4]  = (__bf16)a1[i];   // K = k+4..7
    r[i + 8]  = (__bf16)a2[i];   // K = k+16..19
    r[i + 12] = (__bf16)a3[i];   // K = k+20..23
  }
  return r;
}

__device__ __forceinline__ v16bf pack16(v8bf lo, v8bf hi) {
  v16bf r;
#pragma unroll
  for (int i = 0; i < 8; ++i) { r[i] = lo[i]; r[i + 8] = hi[i]; }
  return r;
}

__global__ __launch_bounds__(256) void moe_router_zero_kernel(
    float* __restrict__ cnt, float* __restrict__ aux, float* __restrict__ sumP) {
  const int t = threadIdx.x;
  if (t < NEXP) { cnt[t] = 0.0f; sumP[t] = 0.0f; }
  if (t == 0) { *aux = 0.0f; }
}

// one-shot fp32 -> bf16 conversion of gate_weight into scratch (L2-resident)
__global__ __launch_bounds__(256) void moe_router_wcvt_kernel(
    const float* __restrict__ W, __bf16* __restrict__ Wbf) {
  const int i = (blockIdx.x * 256 + threadIdx.x) * 4;
  const v4f f = *(const v4f*)(W + i);
  v4bf o;
#pragma unroll
  for (int j = 0; j < 4; ++j) o[j] = (__bf16)f[j];
  *(v4bf*)(Wbf + i) = o;
}

__global__ __launch_bounds__(BLOCK_THREADS) void moe_router_main_kernel(
    const float*  __restrict__ X,      // [NTOK, DMODEL] fp32
    const __bf16* __restrict__ Wbf,    // [NEXP, DMODEL] bf16 (pre-converted)
    float* __restrict__ topk_p,        // [NTOK, 2]
    float* __restrict__ topk_i,        // [NTOK, 2] (indices as float)
    float* __restrict__ cnt_g,         // [NEXP]
    float* __restrict__ sumP_g) {      // [NEXP]
  __shared__ __align__(16) __bf16 Wtile[NEXP * BPITCH];     // ~66.5 KB
  __shared__ float logits[WAVES_PER_BLOCK][ROWS_PER_WAVE * LDS_PITCH];
  __shared__ float s_sum[NEXP];
  __shared__ float s_hist[NEXP];

  const int tid  = threadIdx.x;
  const int wave = tid >> 5;
  const int lane = tid & 31;

  if (tid < NEXP) { s_sum[tid] = 0.0f; s_hist[tid] = 0.0f; }

  const int m0   = (blockIdx.x * WAVES_PER_BLOCK + wave) * ROWS_PER_WAVE;
  const int half = lane >> 4;      // lanes 16..31 hold the K+8 / K+24 halves
  const int sub  = lane & 15;      // row (A) / column (B) within 16
  const int koff = half * 8;

  const float* xr = X + (size_t)(m0 + sub) * DMODEL;
  const unsigned long long wbase = (unsigned long long)(uintptr_t)Wbf;

  v8f c0 = {}, c1 = {}, c2 = {}, c3 = {};

  for (int c = 0; c < DMODEL / KCHUNK; ++c) {
    const int kbase = c * KCHUNK;

    // ---- async-stage this K-chunk of bf16 W into LDS (ASYNCcnt path) ----
#pragma unroll 4
    for (int i = 0; i < GRANULES / BLOCK_THREADS; ++i) {   // 32 granules/thread
      const int id = tid + BLOCK_THREADS * i;              // 0..4095
      const int e  = id >> 6;                              // expert row
      const int g  = id & 63;                              // 8-elem group in chunk
      const unsigned lds_dst = (unsigned)(uintptr_t)&Wtile[e * BPITCH + g * 8];
      const unsigned voff    = (unsigned)((e * DMODEL + kbase + g * 8) * 2);
      asm volatile("global_load_async_to_lds_b128 %0, %1, %2"
                   : : "v"(lds_dst), "v"(voff), "s"(wbase) : "memory");
    }
    asm volatile("s_wait_asynccnt 0x0" : : : "memory");
    __syncthreads();

    // ---- bf16 WMMA over the staged chunk ----
#pragma unroll 4
    for (int kk = 0; kk < KCHUNK; kk += 32) {
      const int ka = kbase + kk + koff;   // global K for A
      const int kl = kk + koff;           // chunk-local K for B
      const v16bf a = cvt16(*(const v4f*)(xr + ka),      *(const v4f*)(xr + ka + 4),
                            *(const v4f*)(xr + ka + 16), *(const v4f*)(xr + ka + 20));
      const v16bf b0 = pack16(*(const v8bf*)&Wtile[(sub)      * BPITCH + kl],
                              *(const v8bf*)&Wtile[(sub)      * BPITCH + kl + 16]);
      const v16bf b1 = pack16(*(const v8bf*)&Wtile[(sub + 16) * BPITCH + kl],
                              *(const v8bf*)&Wtile[(sub + 16) * BPITCH + kl + 16]);
      const v16bf b2 = pack16(*(const v8bf*)&Wtile[(sub + 32) * BPITCH + kl],
                              *(const v8bf*)&Wtile[(sub + 32) * BPITCH + kl + 16]);
      const v16bf b3 = pack16(*(const v8bf*)&Wtile[(sub + 48) * BPITCH + kl],
                              *(const v8bf*)&Wtile[(sub + 48) * BPITCH + kl + 16]);
      c0 = __builtin_amdgcn_wmma_f32_16x16x32_bf16(false, a, false, b0, (short)0, c0, false, false);
      c1 = __builtin_amdgcn_wmma_f32_16x16x32_bf16(false, a, false, b1, (short)0, c1, false, false);
      c2 = __builtin_amdgcn_wmma_f32_16x16x32_bf16(false, a, false, b2, (short)0, c2, false, false);
      c3 = __builtin_amdgcn_wmma_f32_16x16x32_bf16(false, a, false, b3, (short)0, c3, false, false);
    }
    __syncthreads();   // protect Wtile before next chunk's async overwrite
  }

  // ---- spill the 16x64 logits tile to LDS (C layout: vgpr r, lane -> M,N) ----
  {
    float* row = logits[wave];
    const int mbase = half * 8;
#pragma unroll
    for (int r = 0; r < 8; ++r) {
      float* p = row + (mbase + r) * LDS_PITCH + sub;
      p[0]  = c0[r];
      p[16] = c1[r];
      p[32] = c2[r];
      p[48] = c3[r];
    }
  }
  __syncthreads();

  // ---- per-token softmax + top-2 (lanes 0..15 each own one token) ----
  if (lane < 16) {
    const float* v = logits[wave] + lane * LDS_PITCH;
    float maxv = -3.4e38f;
#pragma unroll 8
    for (int e = 0; e < NEXP; ++e) maxv = fmaxf(maxv, v[e]);

    float sum = 0.0f;
    float t1 = -3.4e38f, t2 = -3.4e38f;
    int   i1 = 0, i2 = 0;
    for (int e = 0; e < NEXP; ++e) {
      const float lv = v[e];
      sum += __expf(lv - maxv);
      if (lv > t1)      { t2 = t1; i2 = i1; t1 = lv; i1 = e; }
      else if (lv > t2) { t2 = lv; i2 = e; }
    }
    const float inv = 1.0f / sum;

    for (int e = 0; e < NEXP; ++e)
      atomicAdd(&s_sum[e], __expf(v[e] - maxv) * inv);

    const float p1 = __expf(t1 - maxv) * inv;
    const float p2 = __expf(t2 - maxv) * inv;
    const float ip = 1.0f / (p1 + p2);
    const int tok = m0 + lane;
    topk_p[tok * 2 + 0] = p1 * ip;
    topk_p[tok * 2 + 1] = p2 * ip;
    topk_i[tok * 2 + 0] = (float)i1;
    topk_i[tok * 2 + 1] = (float)i2;
    atomicAdd(&s_hist[i1], 1.0f);
    atomicAdd(&s_hist[i2], 1.0f);
  }
  __syncthreads();

  // ---- one global flush per expert per block ----
  if (tid < NEXP) {
    atomicAdd(&cnt_g[tid],  s_hist[tid]);
    atomicAdd(&sumP_g[tid], s_sum[tid]);
  }
}

__global__ __launch_bounds__(64) void moe_router_aux_kernel(
    const float* __restrict__ cnt_g, const float* __restrict__ sumP_g,
    float* __restrict__ aux) {
  __shared__ float red[NEXP];
  const int t = threadIdx.x;
  red[t] = cnt_g[t] * sumP_g[t];
  __syncthreads();
#pragma unroll
  for (int s = 32; s > 0; s >>= 1) {
    if (t < s) red[t] += red[t + s];
    __syncthreads();
  }
  if (t == 0) {
    // aux = sum_e (cnt/T) * (sumP/T) * E
    *aux = red[0] * ((float)NEXP / ((float)NTOK * (float)NTOK));
  }
}

extern "C" void kernel_launch(void* const* d_in, const int* in_sizes, int n_in,
                              void* d_out, int out_size, void* d_ws, size_t ws_size,
                              hipStream_t stream) {
  const float* X = (const float*)d_in[0];   // [4,4096,2048] fp32
  const float* W = (const float*)d_in[1];   // [64,2048] fp32

  float* out    = (float*)d_out;
  float* topk_p = out;                 // 32768
  float* topk_i = out + 32768;         // 32768 (indices as float)
  float* cnt_g  = out + 65536;         // 64
  float* aux    = out + 65600;         // 1

  float*  sumP_g = (float*)d_ws;                       // 64 floats scratch
  __bf16* Wbf    = (__bf16*)((char*)d_ws + 256);       // 256 KB bf16 copy of W

  moe_router_zero_kernel<<<1, 256, 0, stream>>>(cnt_g, aux, sumP_g);

  // 64*2048 elements / (256 threads * 4 per thread) = 128 blocks
  moe_router_wcvt_kernel<<<128, 256, 0, stream>>>(W, Wbf);

  const int blocks = NTOK / (WAVES_PER_BLOCK * ROWS_PER_WAVE);   // 256
  moe_router_main_kernel<<<blocks, BLOCK_THREADS, 0, stream>>>(
      X, Wbf, topk_p, topk_i, cnt_g, sumP_g);

  moe_router_aux_kernel<<<1, 64, 0, stream>>>(cnt_g, sumP_g, aux);
}
